// Decoder_74165495267514
// MI455X (gfx1250) — matrix operand order repaired and checked by reading, not compiled, not run on hardware
//
#include <hip/hip_runtime.h>
#include <hip/hip_bf16.h>

typedef __bf16 bf16;
typedef __attribute__((ext_vector_type(16))) __bf16 v16bf;
typedef __attribute__((ext_vector_type(8)))  __bf16 v8bf;
typedef __attribute__((ext_vector_type(8)))  float  v8f;

#define HIDN    512
#define ALPHA   40
#define LATENT  128
#define SLEN    120
#define BTOT    8192
#define BT      32      // batch rows per workgroup
#define NWAVE   8       // 256 threads
#define HS      520     // padded LDS row stride (bf16 elements) -> bank-conflict friendly

union V16U { v16bf v; v8bf h[2]; };

__device__ __forceinline__ v8f vzero8() {
  v8f z;
#pragma unroll
  for (int i = 0; i < 8; ++i) z[i] = 0.0f;
  return z;
}

__device__ __forceinline__ v8f wmma_bf16(v16bf a, v16bf b, v8f c) {
  // D = A(16x32 bf16) * B(32x16 bf16) + C(16x16 f32)
  return __builtin_amdgcn_wmma_f32_16x16x32_bf16(false, a, false, b, (short)0, c, false, false);
}

// A-fragment (16x32 bf16) from LDS, row-major with stride HS.
// Lane l<16: rows M=l, K chunks [k0..k0+7] and [k0+16..k0+23];
// lane l>=16: same M, chunks shifted by 8 (per ISA 7.12.2).
__device__ __forceinline__ v16bf ldsA(const bf16* buf, int mbase, int lane, int k0) {
  int row = mbase + (lane & 15);
  int c0  = k0 + ((lane >> 4) << 3);
  V16U a;
  a.h[0] = *(const v8bf*)(buf + row * HS + c0);
  a.h[1] = *(const v8bf*)(buf + row * HS + c0 + 16);
  return a.v;
}

// A-fragment from global bf16 row-major (stride lda)
__device__ __forceinline__ v16bf ldgA(const bf16* __restrict__ buf, int lda, int rowbase,
                                      int lane, int k0) {
  int row = rowbase + (lane & 15);
  int c0  = k0 + ((lane >> 4) << 3);
  V16U a;
  a.h[0] = *(const v8bf*)(buf + row * lda + c0);
  a.h[1] = *(const v8bf*)(buf + row * lda + c0 + 16);
  return a.v;
}

// B-fragment (32x16 bf16): B[k][n] = W[n][k] for GEMM with W^T.
// Lane l holds column n0+(l&15), K chunks [k0+8*(l>>4)..+7] and [k0+16+8*(l>>4)..+7].
__device__ __forceinline__ v16bf ldgB(const bf16* __restrict__ W, int ldw, int n0,
                                      int lane, int k0) {
  int row = n0 + (lane & 15);
  int col = k0 + ((lane >> 4) << 3);
  V16U b;
  b.h[0] = *(const v8bf*)(W + row * ldw + col);
  b.h[1] = *(const v8bf*)(W + row * ldw + col + 16);
  return b.v;
}

__device__ __forceinline__ float sigmoidf_(float x) { return 1.0f / (1.0f + __expf(-x)); }

// O[BT x 512] = A[BT x K] @ W^T + bias  (A,O in LDS bf16; W global bf16 [512 x K])
__device__ __forceinline__ void layer_lds(const bf16* A, const bf16* __restrict__ W,
                                          const float* __restrict__ bias, bf16* O,
                                          int K, int w, int lane) {
  for (int jt = w; jt < 32; jt += NWAVE) {
    int j0 = jt << 4;
    v8f acc0 = vzero8();
    v8f acc1 = vzero8();
    for (int k0 = 0; k0 < K; k0 += 32) {
      v16bf a0 = ldsA(A, 0, lane, k0);
      v16bf a1 = ldsA(A, 16, lane, k0);
      v16bf b  = ldgB(W, K, j0, lane, k0);
      acc0 = wmma_bf16(a0, b, acc0);
      acc1 = wmma_bf16(a1, b, acc1);
    }
    int j   = j0 + (lane & 15);
    float bb = bias[j];
    int mhi = (lane >> 4) << 3;
#pragma unroll
    for (int v = 0; v < 8; ++v) {
      O[(v + mhi) * HS + j]        = (bf16)(acc0[v] + bb);
      O[(16 + v + mhi) * HS + j]   = (bf16)(acc1[v] + bb);
    }
  }
}

__global__ void gru_decode_kernel(const bf16* __restrict__ bcat,   // [1536][1024] = [W_hh | M]
                                  const bf16* __restrict__ h1wb,   // [512][512]
                                  const bf16* __restrict__ h2pb,   // [48][512] (rows>=40 zero)
                                  const bf16* __restrict__ fc1b,   // [512][128]
                                  const bf16* __restrict__ fc2b,   // [512][512]
                                  const bf16* __restrict__ latb,   // [8192][128]
                                  const float* __restrict__ biasblk,
                                  const float* __restrict__ fc1_bias,
                                  const float* __restrict__ fc2_bias,
                                  const float* __restrict__ h1_bias,
                                  const float* __restrict__ h2_bias,
                                  float* __restrict__ out) {
  extern __shared__ bf16 smem[];
  bf16* H = smem;                 // current hidden state (bf16)
  bf16* G = smem + BT * HS;       // next hidden state
  bf16* T = smem + 2 * BT * HS;   // t1 = h_new @ h1^T + h1_b

  const int tid  = threadIdx.x;
  const int lane = tid & 31;
  const int w    = tid >> 5;
  const int b0   = blockIdx.x * BT;

  const float* rb0  = biasblk;             // step-0 r bias  (b_ih+b_hh+gx0)
  const float* zb0  = biasblk + 512;
  const float* nb10 = biasblk + 1024;      // step-0 n bias1 (b_ih_n+gx0_n)
  const float* rbR  = biasblk + 1536;      // steps>=1 (b_ih+b_hh+c)
  const float* zbR  = biasblk + 2048;
  const float* nb1R = biasblk + 2560;
  const float* nb2  = biasblk + 3072;      // b_hh_n

  // y[:,0,:] = x0
  for (int i = tid; i < BT * ALPHA; i += NWAVE * 32) {
    int bm = i / ALPHA, a = i - bm * ALPHA;
    out[(size_t)(b0 + bm) * (SLEN * ALPHA) + a] = (a == 0) ? 16.0f : -16.0f;
  }

  // init1: t0 = latent @ fc1^T + fc1_b -> T
  for (int jt = w; jt < 32; jt += NWAVE) {
    int j0 = jt << 4;
    v8f acc0 = vzero8();
    v8f acc1 = vzero8();
    for (int k0 = 0; k0 < LATENT; k0 += 32) {
      v16bf a0 = ldgA(latb, LATENT, b0, lane, k0);
      v16bf a1 = ldgA(latb, LATENT, b0 + 16, lane, k0);
      v16bf b  = ldgB(fc1b, LATENT, j0, lane, k0);
      acc0 = wmma_bf16(a0, b, acc0);
      acc1 = wmma_bf16(a1, b, acc1);
    }
    int j = j0 + (lane & 15);
    float bb = fc1_bias[j];
    int mhi = (lane >> 4) << 3;
#pragma unroll
    for (int v = 0; v < 8; ++v) {
      T[(v + mhi) * HS + j]      = (bf16)(acc0[v] + bb);
      T[(16 + v + mhi) * HS + j] = (bf16)(acc1[v] + bb);
    }
  }
  __syncthreads();

  // init2: h0 = t0 @ fc2^T + fc2_b -> H
  layer_lds(T, fc2b, fc2_bias, H, HIDN, w, lane);
  __syncthreads();

  for (int s = 0; s < SLEN - 1; ++s) {
    const float* rbS = (s == 0) ? rb0 : rbR;
    const float* zbS = (s == 0) ? zb0 : zbR;
    const float* n1S = (s == 0) ? nb10 : nb1R;

    // phase 1: fused gate GEMMs.  gh from H (K=0..511), gx from T via M (K=512..1023).
    for (int jt = w; jt < 32; jt += NWAVE) {
      int j0 = jt << 4;
      v8f ar0 = vzero8(), ar1 = vzero8();
      v8f az0 = vzero8(), az1 = vzero8();
      v8f agh0 = vzero8(), agh1 = vzero8();
      v8f agx0 = vzero8(), agx1 = vzero8();
      for (int k0 = 0; k0 < HIDN; k0 += 32) {
        v16bf a0 = ldsA(H, 0, lane, k0);
        v16bf a1 = ldsA(H, 16, lane, k0);
        v16bf br = ldgB(bcat, 1024, j0, lane, k0);
        ar0 = wmma_bf16(a0, br, ar0);
        ar1 = wmma_bf16(a1, br, ar1);
        v16bf bz = ldgB(bcat, 1024, 512 + j0, lane, k0);
        az0 = wmma_bf16(a0, bz, az0);
        az1 = wmma_bf16(a1, bz, az1);
        v16bf bn = ldgB(bcat, 1024, 1024 + j0, lane, k0);
        agh0 = wmma_bf16(a0, bn, agh0);
        agh1 = wmma_bf16(a1, bn, agh1);
      }
      if (s > 0) {
        for (int k0 = 0; k0 < HIDN; k0 += 32) {
          v16bf a0 = ldsA(T, 0, lane, k0);
          v16bf a1 = ldsA(T, 16, lane, k0);
          v16bf br = ldgB(bcat, 1024, j0, lane, 512 + k0);
          ar0 = wmma_bf16(a0, br, ar0);
          ar1 = wmma_bf16(a1, br, ar1);
          v16bf bz = ldgB(bcat, 1024, 512 + j0, lane, 512 + k0);
          az0 = wmma_bf16(a0, bz, az0);
          az1 = wmma_bf16(a1, bz, az1);
          v16bf bn = ldgB(bcat, 1024, 1024 + j0, lane, 512 + k0);
          agx0 = wmma_bf16(a0, bn, agx0);
          agx1 = wmma_bf16(a1, bn, agx1);
        }
      }
      int j = j0 + (lane & 15);
      float rb = rbS[j], zb = zbS[j], n1 = n1S[j], n2v = nb2[j];
      int mhi = (lane >> 4) << 3;
#pragma unroll
      for (int v = 0; v < 8; ++v) {
        {
          int row = v + mhi;
          float hold = (float)H[row * HS + j];
          float r  = sigmoidf_(ar0[v] + rb);
          float z  = sigmoidf_(az0[v] + zb);
          float nn = tanhf(agx0[v] + n1 + r * (agh0[v] + n2v));
          G[row * HS + j] = (bf16)((1.0f - z) * nn + z * hold);
        }
        {
          int row = 16 + v + mhi;
          float hold = (float)H[row * HS + j];
          float r  = sigmoidf_(ar1[v] + rb);
          float z  = sigmoidf_(az1[v] + zb);
          float nn = tanhf(agx1[v] + n1 + r * (agh1[v] + n2v));
          G[row * HS + j] = (bf16)((1.0f - z) * nn + z * hold);
        }
      }
    }
    __syncthreads();

    // phase 2: t1 = h_new @ h1^T + h1_b -> T
    layer_lds(G, h1wb, h1_bias, T, HIDN, w, lane);
    __syncthreads();

    // phase 3: pred = t1 @ h2pad^T + h2_b -> out[:, s+1, 0:40]
    if (w < 6) {
      int jt = w >> 1, mt = w & 1;
      int j0 = jt << 4;
      v8f acc = vzero8();
      for (int k0 = 0; k0 < HIDN; k0 += 32) {
        v16bf a = ldsA(T, mt << 4, lane, k0);
        v16bf b = ldgB(h2pb, HIDN, j0, lane, k0);
        acc = wmma_bf16(a, b, acc);
      }
      int n = j0 + (lane & 15);
      if (n < ALPHA) {
        float bb = h2_bias[n];
        int mhi = (lane >> 4) << 3;
#pragma unroll
        for (int v = 0; v < 8; ++v) {
          int bm = b0 + (mt << 4) + v + mhi;
          out[(size_t)bm * (SLEN * ALPHA) + (size_t)(s + 1) * ALPHA + n] = acc[v] + bb;
        }
      }
    }

    // swap current/next hidden buffers
    bf16* tmpp = H; H = G; G = tmpp;
    __syncthreads();
  }
}

// ---------------- prep kernels ----------------

__global__ void k_cvt(const float* __restrict__ s, bf16* __restrict__ d, int n) {
  int i = blockIdx.x * blockDim.x + threadIdx.x;
  if (i < n) d[i] = (bf16)s[i];
}

__global__ void k_h2pad(const float* __restrict__ h2w, bf16* __restrict__ d) {
  int i = blockIdx.x * blockDim.x + threadIdx.x;
  if (i < 48 * HIDN) {
    int r = i >> 9, c = i & 511;
    d[i] = (r < ALPHA) ? (bf16)h2w[r * HIDN + c] : (bf16)0.0f;
  }
}

// Bcat[j][0:512] = W_hh[j][:]; Bcat[j][512:1024] = M[j][:] = sum_a W_ih[j][a]*h2_w[a][:]
__global__ void k_bcat(const float* __restrict__ Whh, const float* __restrict__ Wih,
                       const float* __restrict__ h2w, bf16* __restrict__ d) {
  int i = blockIdx.x * blockDim.x + threadIdx.x;
  if (i >= 1536 * 1024) return;
  int j = i >> 10, k = i & 1023;
  float v;
  if (k < HIDN) {
    v = Whh[j * HIDN + k];
  } else {
    int kk = k - HIDN;
    float acc = 0.0f;
    for (int a = 0; a < ALPHA; ++a) acc += Wih[j * ALPHA + a] * h2w[a * HIDN + kk];
    v = acc;
  }
  d[i] = (bf16)v;
}

// Fold gx0 (x0 @ W_ih^T), c (h2_b @ W_ih^T) and b_ih/b_hh into per-gate bias vectors.
__global__ void k_bias(const float* __restrict__ Wih, const float* __restrict__ bih,
                       const float* __restrict__ bhh, const float* __restrict__ h2b,
                       float* __restrict__ bb) {
  int jj = blockIdx.x * blockDim.x + threadIdx.x;
  if (jj >= 1536) return;
  float gx0 = 16.0f * Wih[jj * ALPHA];
  for (int a = 1; a < ALPHA; ++a) gx0 -= 16.0f * Wih[jj * ALPHA + a];
  float c = 0.0f;
  for (int a = 0; a < ALPHA; ++a) c += Wih[jj * ALPHA + a] * h2b[a];
  int g = jj >> 9, j = jj & 511;
  if (g == 0) {
    bb[j]        = bih[jj] + bhh[jj] + gx0;
    bb[1536 + j] = bih[jj] + bhh[jj] + c;
  } else if (g == 1) {
    bb[512 + j]  = bih[jj] + bhh[jj] + gx0;
    bb[2048 + j] = bih[jj] + bhh[jj] + c;
  } else {
    bb[1024 + j] = bih[jj] + gx0;
    bb[2560 + j] = bih[jj] + c;
    bb[3072 + j] = bhh[jj];
  }
}

extern "C" void kernel_launch(void* const* d_in, const int* in_sizes, int n_in,
                              void* d_out, int out_size, void* d_ws, size_t ws_size,
                              hipStream_t stream) {
  const float* latent = (const float*)d_in[0];
  const float* fc1_w  = (const float*)d_in[1];
  const float* fc1_b  = (const float*)d_in[2];
  const float* fc2_w  = (const float*)d_in[3];
  const float* fc2_b  = (const float*)d_in[4];
  const float* W_ih   = (const float*)d_in[5];
  const float* W_hh   = (const float*)d_in[6];
  const float* b_ih   = (const float*)d_in[7];
  const float* b_hh   = (const float*)d_in[8];
  const float* h1_w   = (const float*)d_in[9];
  const float* h1_b   = (const float*)d_in[10];
  const float* h2_w   = (const float*)d_in[11];
  const float* h2_b   = (const float*)d_in[12];
  float* out = (float*)d_out;

  bf16* bcat = (bf16*)d_ws;
  bf16* h1wb = bcat + (size_t)1536 * 1024;
  bf16* h2pb = h1wb + (size_t)512 * 512;
  bf16* fc1b = h2pb + (size_t)48 * 512;
  bf16* fc2b = fc1b + (size_t)512 * 128;
  bf16* latb = fc2b + (size_t)512 * 512;
  float* bb  = (float*)(latb + (size_t)BTOT * LATENT);

  const int T256 = 256;
  k_cvt<<<(512 * 512 + T256 - 1) / T256, T256, 0, stream>>>(h1_w, h1wb, 512 * 512);
  k_cvt<<<(512 * 128 + T256 - 1) / T256, T256, 0, stream>>>(fc1_w, fc1b, 512 * 128);
  k_cvt<<<(512 * 512 + T256 - 1) / T256, T256, 0, stream>>>(fc2_w, fc2b, 512 * 512);
  k_cvt<<<(BTOT * LATENT + T256 - 1) / T256, T256, 0, stream>>>(latent, latb, BTOT * LATENT);
  k_h2pad<<<(48 * HIDN + T256 - 1) / T256, T256, 0, stream>>>(h2_w, h2pb);
  k_bcat<<<(1536 * 1024 + T256 - 1) / T256, T256, 0, stream>>>(W_hh, W_ih, h2_w, bcat);
  k_bias<<<(1536 + T256 - 1) / T256, T256, 0, stream>>>(W_ih, b_ih, b_hh, h2_b, bb);

  const int shmem = 3 * BT * HS * (int)sizeof(bf16);  // 99,840 bytes
  (void)hipFuncSetAttribute((const void*)gru_decode_kernel,
                            hipFuncAttributeMaxDynamicSharedMemorySize, shmem);
  gru_decode_kernel<<<BTOT / BT, NWAVE * 32, shmem, stream>>>(
      bcat, h1wb, h2pb, fc1b, fc2b, latb, bb, fc1_b, fc2_b, h1_b, h2_b, out);
}
